// InteractionBlock_2774548873996
// MI455X (gfx1250) — matrix operand order repaired and checked by reading, not compiled, not run on hardware
//
#include <hip/hip_runtime.h>

// Problem constants
constexpr int FDIM   = 128;  // feature dim
constexpr int NNB    = 64;   // neighbors per atom
constexpr int SDIM   = 50;   // filter basis size
constexpr int AATOMS = 1024; // atoms per batch
constexpr int BA     = 8192; // B*A rows
constexpr int WSTR   = 136;  // LDS stride (halves) for 128-wide tiles (16B aligned, conflict-free)
constexpr int ESTR   = 72;   // LDS stride (halves) for 64-wide tiles
constexpr int APB    = 4;    // atoms per block in edge kernel (amortize G_W staging)

typedef __attribute__((ext_vector_type(16))) _Float16 v16h;
typedef __attribute__((ext_vector_type(8)))  float    v8f;

__device__ __forceinline__ float ssp(float x) {
    // softplus(x) - log(2), numerically stable
    return fmaxf(x, 0.f) + log1pf(__expf(-fabsf(x))) - 0.6931471805599453f;
}

// A-fragment (16x32 f16, M x K): lane l holds row M = l%16.
// halves e=0..7  -> K = k0 + 8*(l/16) + e        (contiguous 16B)
// halves e=8..15 -> K = k0 + 8*(l/16) + 8 + e    (contiguous 16B at +16 halves)
__device__ __forceinline__ v16h load_a_frag(const _Float16* lds, int stride,
                                            int row0, int k0, int lane) {
    int row = row0 + (lane & 15);
    int kb  = k0 + ((lane >> 4) << 3);
    const _Float16* p = lds + row * stride + kb;
    union { v16h v; uint4 u[2]; } r;
    r.u[0] = *(const uint4*)(p);
    r.u[1] = *(const uint4*)(p + 16);
    return r.v;
}

// B-fragment (32x16 f16, K x N) from K-contiguous (transposed) LDS: lane l holds
// col N = l%16; halves e=0..7 -> K = k0 + 8*(l/16) + e, e=8..15 -> K = k0 + 8*(l/16) + 8 + e
__device__ __forceinline__ v16h load_b_frag(const _Float16* lds, int stride,
                                            int col0, int k0, int lane) {
    int col = col0 + (lane & 15);
    int kb  = k0 + ((lane >> 4) << 3);
    const _Float16* p = lds + col * stride + kb;
    union { v16h v; uint4 u[2]; } r;
    r.u[0] = *(const uint4*)(p);
    r.u[1] = *(const uint4*)(p + 16);
    return r.v;
}

// Y = [optional ssp]( ssp(X) @ W + bias [+ addT] [+ mulV*mulT] )
// X:[M,128] f32, W:[128,128] f32 row-major (k,n). 64-row tile per block, 256 thr.
__global__ __launch_bounds__(256) void dense128_kernel(
    const float* __restrict__ X, const float* __restrict__ W,
    const float* __restrict__ bias, const float* __restrict__ addT,
    const float* __restrict__ mulV, const float* __restrict__ mulT,
    float* __restrict__ Y, int outAct)
{
    __shared__ __align__(16) _Float16 Wt[FDIM * WSTR];  // transposed: Wt[n][k]
    __shared__ __align__(16) _Float16 Xs[64 * WSTR];    // ssp(X) tile f16

    const int tid  = threadIdx.x;
    const int row0 = blockIdx.x * 64;

    // Stage W (transpose to K-contiguous f16). Coalesced global reads.
    #pragma unroll 4
    for (int i = 0; i < 64; ++i) {
        int idx = i * 256 + tid;
        int k = idx >> 7, n = idx & 127;
        Wt[n * WSTR + k] = (_Float16)W[idx];
    }
    // Stage activated input tile (float4 loads).
    #pragma unroll
    for (int i = 0; i < 8; ++i) {
        int idx4 = i * 256 + tid;
        int r = idx4 >> 5, c4 = (idx4 & 31) * 4;
        const float4 v = *(const float4*)(X + (size_t)(row0 + r) * FDIM + c4);
        _Float16* d = Xs + r * WSTR + c4;
        d[0] = (_Float16)ssp(v.x); d[1] = (_Float16)ssp(v.y);
        d[2] = (_Float16)ssp(v.z); d[3] = (_Float16)ssp(v.w);
    }
    __syncthreads();

    const int lane = tid & 31, wid = tid >> 5;
    const int mT = wid & 3;    // M tile 0..3
    const int nG = wid >> 2;   // N group 0..1 (4 tiles each)

    v8f acc0 = {}, acc1 = {}, acc2 = {}, acc3 = {};
    #pragma unroll
    for (int kk = 0; kk < 4; ++kk) {
        const int k0 = kk * 32;
        v16h a  = load_a_frag(Xs, WSTR, mT * 16, k0, lane);
        v16h b0 = load_b_frag(Wt, WSTR, (nG * 4 + 0) * 16, k0, lane);
        v16h b1 = load_b_frag(Wt, WSTR, (nG * 4 + 1) * 16, k0, lane);
        v16h b2 = load_b_frag(Wt, WSTR, (nG * 4 + 2) * 16, k0, lane);
        v16h b3 = load_b_frag(Wt, WSTR, (nG * 4 + 3) * 16, k0, lane);
        acc0 = __builtin_amdgcn_wmma_f32_16x16x32_f16(false, a, false, b0, (short)0, acc0, false, false);
        acc1 = __builtin_amdgcn_wmma_f32_16x16x32_f16(false, a, false, b1, (short)0, acc1, false, false);
        acc2 = __builtin_amdgcn_wmma_f32_16x16x32_f16(false, a, false, b2, (short)0, acc2, false, false);
        acc3 = __builtin_amdgcn_wmma_f32_16x16x32_f16(false, a, false, b3, (short)0, acc3, false, false);
    }

    // Epilogue directly from accumulator layout: VGPR j -> row j + 8*(lane/16), col = lane%16.
    v8f accs[4] = {acc0, acc1, acc2, acc3};
    const int colBase = lane & 15;
    const int rAdd = (lane >> 4) * 8;
    #pragma unroll
    for (int t = 0; t < 4; ++t) {
        const int col = (nG * 4 + t) * 16 + colBase;
        const float bv = bias ? bias[col] : 0.f;
        const float mv = mulV ? mulV[col] : 0.f;
        union { v8f v; float f[8]; } u; u.v = accs[t];
        #pragma unroll
        for (int j = 0; j < 8; ++j) {
            const size_t rowG = (size_t)(row0 + mT * 16 + rAdd + j);
            float val = u.f[j] + bv;
            if (addT) val += addT[rowG * FDIM + col];
            if (mulT) val += mv * mulT[rowG * FDIM + col];
            if (outAct) val = ssp(val);
            Y[rowG * FDIM + col] = val;
        }
    }
}

// Per-atom continuous-filter conv: T[a,:] = y[a,:] + sum_n (f_ij[a,n,:]@G_W) * y[nb] * mask
// APB atoms per block; G_W staged once per block.
__global__ __launch_bounds__(256) void edge_kernel(
    const float* __restrict__ y, const float* __restrict__ f_ij,
    const float* __restrict__ G_W, const int* __restrict__ neighbors,
    const float* __restrict__ nmask, float* __restrict__ T)
{
    __shared__ __align__(16) _Float16 Gt[FDIM * ESTR]; // Gt[f][s] (S padded to 64)
    __shared__ __align__(16) _Float16 Fs[NNB * ESTR];  // Fs[n][s]
    __shared__ int   nbs[NNB];
    __shared__ float msk[NNB];
    __shared__ float accF[FDIM];

    const int tid   = threadIdx.x;
    const int atom0 = blockIdx.x * APB;
    const int lane  = tid & 31, wid = tid >> 5;
    const int mT = wid & 3;    // neighbor-row tile
    const int nG = wid >> 2;   // feature group (4 x 16 cols)
    const int colBase = lane & 15;
    const int rAdd = (lane >> 4) * 8;

    // Stage G_W transposed once (L2-resident, tiny) with zero-pad S 50 -> 64.
    #pragma unroll 4
    for (int i = 0; i < 32; ++i) {
        int idx = i * 256 + tid;
        int s = idx >> 7, f = idx & 127;
        float v = (s < SDIM) ? G_W[s * FDIM + f] : 0.f;
        Gt[f * ESTR + s] = (_Float16)v;
    }
    // Prefetch first atom's f_ij block (~12.8 KB) into cache.
    __builtin_prefetch(f_ij + (size_t)atom0 * NNB * SDIM + (size_t)tid * 16, 0, 3);

    for (int ai = 0; ai < APB; ++ai) {
        const int atom = atom0 + ai;
        const int b    = atom >> 10; // batch index

        // Stage this atom's f_ij [64 x 50 -> 64] as f16.
        #pragma unroll 4
        for (int i = 0; i < 16; ++i) {
            int idx = i * 256 + tid;
            int n = idx >> 6, s = idx & 63;
            float v = (s < SDIM) ? f_ij[((size_t)atom * NNB + n) * SDIM + s] : 0.f;
            Fs[n * ESTR + s] = (_Float16)v;
        }
        if (tid < NNB) {
            nbs[tid] = neighbors[(size_t)atom * NNB + tid];
            msk[tid] = nmask[(size_t)atom * NNB + tid];
        }
        if (tid < FDIM) accF[tid] = 0.f;
        // Prefetch next atom's f_ij while this atom computes.
        if (ai + 1 < APB)
            __builtin_prefetch(f_ij + (size_t)(atom + 1) * NNB * SDIM + (size_t)tid * 16, 0, 3);
        __syncthreads();

        v8f acc0 = {}, acc1 = {}, acc2 = {}, acc3 = {};
        #pragma unroll
        for (int kk = 0; kk < 2; ++kk) {
            const int k0 = kk * 32;
            v16h a  = load_a_frag(Fs, ESTR, mT * 16, k0, lane);
            v16h b0 = load_b_frag(Gt, ESTR, (nG * 4 + 0) * 16, k0, lane);
            v16h b1 = load_b_frag(Gt, ESTR, (nG * 4 + 1) * 16, k0, lane);
            v16h b2 = load_b_frag(Gt, ESTR, (nG * 4 + 2) * 16, k0, lane);
            v16h b3 = load_b_frag(Gt, ESTR, (nG * 4 + 3) * 16, k0, lane);
            acc0 = __builtin_amdgcn_wmma_f32_16x16x32_f16(false, a, false, b0, (short)0, acc0, false, false);
            acc1 = __builtin_amdgcn_wmma_f32_16x16x32_f16(false, a, false, b1, (short)0, acc1, false, false);
            acc2 = __builtin_amdgcn_wmma_f32_16x16x32_f16(false, a, false, b2, (short)0, acc2, false, false);
            acc3 = __builtin_amdgcn_wmma_f32_16x16x32_f16(false, a, false, b3, (short)0, acc3, false, false);
        }

        // Gather-multiply-reduce straight from accumulator layout; y rows are L2-resident.
        union { v8f v; float f[8]; } u0, u1, u2, u3;
        u0.v = acc0; u1.v = acc1; u2.v = acc2; u3.v = acc3;
        float part0 = 0.f, part1 = 0.f, part2 = 0.f, part3 = 0.f;
        #pragma unroll
        for (int j = 0; j < 8; ++j) {
            const int n = mT * 16 + rAdd + j;
            const float m = msk[n];
            const float* yrow = y + ((size_t)b * AATOMS + nbs[n]) * FDIM;
            part0 += u0.f[j] * m * yrow[(nG * 4 + 0) * 16 + colBase];
            part1 += u1.f[j] * m * yrow[(nG * 4 + 1) * 16 + colBase];
            part2 += u2.f[j] * m * yrow[(nG * 4 + 2) * 16 + colBase];
            part3 += u3.f[j] * m * yrow[(nG * 4 + 3) * 16 + colBase];
        }
        atomicAdd(&accF[(nG * 4 + 0) * 16 + colBase], part0);
        atomicAdd(&accF[(nG * 4 + 1) * 16 + colBase], part1);
        atomicAdd(&accF[(nG * 4 + 2) * 16 + colBase], part2);
        atomicAdd(&accF[(nG * 4 + 3) * 16 + colBase], part3);
        __syncthreads();

        if (tid < FDIM)
            T[(size_t)atom * FDIM + tid] = y[(size_t)atom * FDIM + tid] + accF[tid];
        __syncthreads(); // protect Fs/nbs/msk/accF before next atom's staging
    }
}

extern "C" void kernel_launch(void* const* d_in, const int* in_sizes, int n_in,
                              void* d_out, int out_size, void* d_ws, size_t ws_size,
                              hipStream_t stream) {
    const float* x         = (const float*)d_in[0];
    /* d_in[1] = r_ij : unused by the reference */
    const int*   neighbors = (const int*)  d_in[2];
    const float* nmask     = (const float*)d_in[3];
    const float* f_ij      = (const float*)d_in[4];
    const float* in2f_W    = (const float*)d_in[5];
    const float* in2f_b    = (const float*)d_in[6];
    const float* G_W       = (const float*)d_in[7];
    const float* res_Ws    = (const float*)d_in[8];
    const float* res_bs    = (const float*)d_in[9];
    const float* dense_W   = (const float*)d_in[10];
    const float* dense_b   = (const float*)d_in[11];
    const float* mask_p    = (const float*)d_in[12];
    float* out = (float*)d_out;

    const size_t TN = (size_t)BA * FDIM;
    float* y  = (float*)d_ws;   // ssp(ssp(x)@W+b)
    float* t  = y + TN;         // y + conv
    float* h  = t + TN;         // scratch / y4
    float* h2 = h + TN;         // scratch

    dim3 blk(256);
    dim3 gD(BA / 64);
    dim3 gE(BA / APB);

    // 1) y = ssp( ssp(x) @ in2f_W + in2f_b )
    dense128_kernel<<<gD, blk, 0, stream>>>(x, in2f_W, in2f_b, nullptr, nullptr, nullptr, y, 1);
    // 2) t = y + masked sum_n (f_ij @ G_W) * y[neighbors]
    edge_kernel<<<gE, blk, 0, stream>>>(y, f_ij, G_W, neighbors, nmask, t);
    // 3-5) residual chain: h_i = ssp(h_{i-1}) @ W_i + b_i ; fold y-add into layer 3
    dense128_kernel<<<gD, blk, 0, stream>>>(t,  res_Ws,                   res_bs,            nullptr, nullptr, nullptr, h,  0);
    dense128_kernel<<<gD, blk, 0, stream>>>(h,  res_Ws + FDIM * FDIM,     res_bs + FDIM,     nullptr, nullptr, nullptr, h2, 0);
    dense128_kernel<<<gD, blk, 0, stream>>>(h2, res_Ws + 2 * FDIM * FDIM, res_bs + 2 * FDIM, t,       nullptr, nullptr, h,  0);
    // 6) out = ssp(y4) @ dense_W + dense_b + mask_p * x
    dense128_kernel<<<gD, blk, 0, stream>>>(h, dense_W, dense_b, nullptr, mask_p, x, out, 0);
}
